// AFTLocalAutoregressive_90211493085408
// MI455X (gfx1250) — matrix operand order repaired and checked by reading, not compiled, not run on hardware
//
#include <hip/hip_runtime.h>
#include <hip/hip_bf16.h>

// ---------------------------------------------------------------------------
// AFT-Local autoregressive layer for MI455X (gfx1250, wave32, WMMA).
// T=2048, B=2, D=512, S=32. GEMMs (M=4096, N=K=512) run on the matrix pipes
// via v_wmma_f32_16x16x32_bf16 with double-buffered LDS fed by
// global_load_async_to_lds_b128 (ASYNCcnt-tracked, no VGPR staging).
// ---------------------------------------------------------------------------

#define T_LEN 2048
#define BATCH 2
#define DMODEL 512
#define WIN 32
#define NCOL (BATCH * DMODEL)   // 1024 columns (b,d)
#define NCHUNK 16
#define CHUNK (T_LEN / NCHUNK)  // 128

typedef __attribute__((ext_vector_type(16))) __bf16 v16bf;
typedef __attribute__((ext_vector_type(8)))  float  v8f;

// ---------------------------------------------------------------------------
// 0) fp32 -> bf16 conversion (activations + weights), 4 elems/thread.
// ---------------------------------------------------------------------------
__global__ void f32_to_bf16(const float* __restrict__ src, __bf16* __restrict__ dst) {
    int i = (blockIdx.x * blockDim.x + threadIdx.x) * 4;
    float4 f = *(const float4*)(src + i);
    dst[i + 0] = (__bf16)f.x;
    dst[i + 1] = (__bf16)f.y;
    dst[i + 2] = (__bf16)f.z;
    dst[i + 3] = (__bf16)f.w;
}

// ---------------------------------------------------------------------------
// 1) pbmax_relu[t] = max(0, max_r pos_bias[r, t])
// ---------------------------------------------------------------------------
__global__ void pb_colmax(const float* __restrict__ pb, float* __restrict__ pbm) {
    int t = blockIdx.x * blockDim.x + threadIdx.x;
    float m = -3.4e38f;
    for (int r = 0; r < T_LEN; ++r)
        m = fmaxf(m, pb[(size_t)r * T_LEN + t]);
    pbm[t] = fmaxf(m, 0.0f);
}

// ---------------------------------------------------------------------------
// 2) C[m,n] = sum_k A[m,k] * W[n,k] + bias[n]   (A,W in bf16; C f32)
//    Block = 64x64 tile, 128 threads (4 waves), each wave a 32x32 quadrant.
//    Double-buffered LDS filled by async global->LDS B128 copies.
// ---------------------------------------------------------------------------
#define GM 4096
#define GN 512
#define GK 512
#define BM 64
#define BN 64
#define BK 32
#define PADK 40   // 80B row stride: 16B-aligned chunks, conflict-free ds reads

__device__ __forceinline__ void async_b128_to_lds(unsigned lds_off, const void* gptr) {
    asm volatile("global_load_async_to_lds_b128 %0, %1, off"
                 :: "v"(lds_off), "v"(gptr) : "memory");
}
__device__ __forceinline__ void wait_async0() {
    asm volatile("s_wait_asynccnt 0x0" ::: "memory");
}

__global__ __launch_bounds__(128)
void gemm_bf16_wmma(const __bf16* __restrict__ A, const __bf16* __restrict__ W,
                    const float* __restrict__ bias, float* __restrict__ C) {
    __shared__ __bf16 sA[2][BM][PADK];
    __shared__ __bf16 sB[2][BN][PADK];

    const int m0 = blockIdx.y * BM;
    const int n0 = blockIdx.x * BN;

    const int lane = threadIdx.x & 31;
    const int wave = threadIdx.x >> 5;
    const int wm = (wave >> 1) * 32;
    const int wn = (wave & 1) * 32;
    const int r    = lane & 15;
    const int half = lane >> 4;

    // Async tile fetch: 64 rows x 32 bf16 = 256 chunks of 16B per matrix;
    // 128 threads -> 2 chunks each per matrix.
    auto issue_tile = [&](int buf, int k0) {
        #pragma unroll
        for (int rr = 0; rr < 2; ++rr) {
            const int c   = threadIdx.x + rr * 128;   // 0..255
            const int row = c >> 2;
            const int cc  = (c & 3) * 8;              // bf16 column offset
            async_b128_to_lds((unsigned)(uintptr_t)&sA[buf][row][cc],
                              A + (size_t)(m0 + row) * GK + k0 + cc);
            async_b128_to_lds((unsigned)(uintptr_t)&sB[buf][row][cc],
                              W + (size_t)(n0 + row) * GK + k0 + cc);
        }
    };

    v8f acc[2][2] = {};

    issue_tile(0, 0);
    int buf = 0;
    for (int k0 = 0; k0 < GK; k0 += BK) {
        wait_async0();        // own async copies into `buf` done
        __syncthreads();      // everyone's copies visible; prev reads finished
        if (k0 + BK < GK)
            issue_tile(buf ^ 1, k0 + BK);   // overlap next tile with compute

        // A fragments (16x32 bf16): lanes0-15 K 0..7/16..23, lanes16-31 K 8..15/24..31.
        v16bf af[2];
        #pragma unroll
        for (int tm = 0; tm < 2; ++tm) {
            const int mrow = wm + 16 * tm + r;
            #pragma unroll
            for (int v = 0; v < 8; ++v) {
                const int kb = ((v < 4) ? (2 * v) : (8 + 2 * v)) + 8 * half;
                af[tm][2 * v]     = sA[buf][mrow][kb];
                af[tm][2 * v + 1] = sA[buf][mrow][kb + 1];
            }
        }
        // B fragments (32x16 bf16): lane = N; lanes0-15 K 0..7/16..23, lanes16-31 K 8..15/24..31.
        v16bf bfr[2];
        #pragma unroll
        for (int tn = 0; tn < 2; ++tn) {
            const int nrow = wn + 16 * tn + r;
            #pragma unroll
            for (int v = 0; v < 8; ++v) {
                const int kb = ((v < 4) ? (2 * v) : (8 + 2 * v)) + 8 * half;
                bfr[tn][2 * v]     = sB[buf][nrow][kb];
                bfr[tn][2 * v + 1] = sB[buf][nrow][kb + 1];
            }
        }
        #pragma unroll
        for (int tm = 0; tm < 2; ++tm)
            #pragma unroll
            for (int tn = 0; tn < 2; ++tn)
                acc[tm][tn] = __builtin_amdgcn_wmma_f32_16x16x32_bf16(
                    false, af[tm], false, bfr[tn], (short)0, acc[tm][tn],
                    false, false);
        buf ^= 1;
    }

    // Epilogue: f32 C layout — VGPR v: lanes0-15 -> M=v, lanes16-31 -> M=8+v; N=lane%16.
    #pragma unroll
    for (int tm = 0; tm < 2; ++tm) {
        #pragma unroll
        for (int tn = 0; tn < 2; ++tn) {
            const int ng = n0 + wn + 16 * tn + r;
            const float bb = bias[ng];
            #pragma unroll
            for (int v = 0; v < 8; ++v) {
                const int mg = m0 + wm + 16 * tm + v + 8 * half;
                C[(size_t)mg * GN + ng] = acc[tm][tn][v] + bb;
            }
        }
    }
}

// ---------------------------------------------------------------------------
// 3) max_logit[d] = max_t (k[t,0,d] + pbmax_relu[t])
// ---------------------------------------------------------------------------
__global__ void max_logit_kernel(const float* __restrict__ k,
                                 const float* __restrict__ pbm,
                                 float* __restrict__ ml) {
    int d = blockIdx.x * blockDim.x + threadIdx.x;
    float m = -3.4e38f;
    for (int t = 0; t < T_LEN; ++t)
        m = fmaxf(m, k[(size_t)t * NCOL + d] + pbm[t]);
    ml[d] = m;
}

// ---------------------------------------------------------------------------
// 4) Chunked cumulative sums of ek = exp(k - ml) and ek*v over T, per column.
// ---------------------------------------------------------------------------
__global__ void scan_pass1(const float* __restrict__ k, const float* __restrict__ v,
                           const float* __restrict__ ml,
                           float* __restrict__ pnum, float* __restrict__ pden) {
    int col = blockIdx.y * blockDim.x + threadIdx.x;
    int c   = blockIdx.x;
    float mld = ml[col & (DMODEL - 1)];
    float sn = 0.f, sd = 0.f;
    for (int t = c * CHUNK; t < (c + 1) * CHUNK; ++t) {
        float e = __expf(k[(size_t)t * NCOL + col] - mld);
        sd += e;
        sn += e * v[(size_t)t * NCOL + col];
    }
    pnum[c * NCOL + col] = sn;
    pden[c * NCOL + col] = sd;
}

__global__ void scan_pass2(float* __restrict__ pnum, float* __restrict__ pden) {
    int col = blockIdx.x * blockDim.x + threadIdx.x;
    float rn = 0.f, rd = 0.f;
    #pragma unroll
    for (int c = 0; c < NCHUNK; ++c) {
        float tn = pnum[c * NCOL + col];
        float td = pden[c * NCOL + col];
        pnum[c * NCOL + col] = rn;
        pden[c * NCOL + col] = rd;
        rn += tn;
        rd += td;
    }
}

__global__ void scan_pass3(const float* __restrict__ k, const float* __restrict__ v,
                           const float* __restrict__ ml,
                           const float* __restrict__ pnum, const float* __restrict__ pden,
                           float* __restrict__ cs_num, float* __restrict__ cs_den) {
    int col = blockIdx.y * blockDim.x + threadIdx.x;
    int c   = blockIdx.x;
    float mld = ml[col & (DMODEL - 1)];
    float rn = pnum[c * NCOL + col];
    float rd = pden[c * NCOL + col];
    for (int t = c * CHUNK; t < (c + 1) * CHUNK; ++t) {
        float e = __expf(k[(size_t)t * NCOL + col] - mld);
        rd += e;
        rn += e * v[(size_t)t * NCOL + col];
        cs_den[(size_t)t * NCOL + col] = rd;
        cs_num[(size_t)t * NCOL + col] = rn;
    }
}

// ---------------------------------------------------------------------------
// 5) Prefix + 32-tap local window + sigmoid gate -> y (bf16, feeds final GEMM)
// ---------------------------------------------------------------------------
__global__ __launch_bounds__(512)
void aft_combine(const float* __restrict__ q, const float* __restrict__ k,
                 const float* __restrict__ v,
                 const float* __restrict__ cs_num, const float* __restrict__ cs_den,
                 const float* __restrict__ pb, const float* __restrict__ ml,
                 __bf16* __restrict__ y) {
    const int t = blockIdx.x;
    const int b = blockIdx.y;
    const int d = threadIdx.x;
    const int col = b * DMODEL + d;
    const float mld = ml[d];

    float num = 0.f, den = 0.f;
    if (t >= WIN) {
        num = cs_num[(size_t)(t - WIN) * NCOL + col];
        den = cs_den[(size_t)(t - WIN) * NCOL + col];
    }
    const int j0 = (t >= WIN - 1) ? 0 : (WIN - 1 - t);
    for (int j = j0; j < WIN; ++j) {
        const int idx = t - WIN + 1 + j;
        const float pbv = pb[(size_t)t * T_LEN + idx];  // wave-uniform -> scalar
        const float w = __expf(k[(size_t)idx * NCOL + col] + pbv - mld);
        den += w;
        num += w * v[(size_t)idx * NCOL + col];
    }
    const float qv = q[(size_t)t * NCOL + col];
    const float sg = 1.0f / (1.0f + __expf(-qv));
    y[(size_t)t * NCOL + col] = (__bf16)(sg * num / den);
}

// ---------------------------------------------------------------------------
// Launch
// ---------------------------------------------------------------------------
extern "C" void kernel_launch(void* const* d_in, const int* in_sizes, int n_in,
                              void* d_out, int out_size, void* d_ws, size_t ws_size,
                              hipStream_t stream) {
    (void)in_sizes; (void)n_in; (void)out_size; (void)ws_size;

    const float* query = (const float*)d_in[0];
    const float* key_  = (const float*)d_in[1];
    const float* value = (const float*)d_in[2];
    const float* Wq    = (const float*)d_in[3];
    const float* bq    = (const float*)d_in[4];
    const float* Wk    = (const float*)d_in[5];
    const float* bk    = (const float*)d_in[6];
    const float* Wv    = (const float*)d_in[7];
    const float* bv    = (const float*)d_in[8];
    const float* pb    = (const float*)d_in[9];
    const float* Wo    = (const float*)d_in[10];
    const float* bo    = (const float*)d_in[11];
    float* out = (float*)d_out;

    const size_t MT = (size_t)T_LEN * NCOL;   // 2M elems per activation tensor
    const size_t WT = (size_t)DMODEL * DMODEL;

    char* w = (char*)d_ws;
    float*  q      = (float*)w;  w += MT * 4;
    float*  kbuf   = (float*)w;  w += MT * 4;
    float*  vbuf   = (float*)w;  w += MT * 4;
    float*  cs_num = (float*)w;  w += MT * 4;
    float*  cs_den = (float*)w;  w += MT * 4;
    __bf16* qx     = (__bf16*)w; w += MT * 2;
    __bf16* kx     = (__bf16*)w; w += MT * 2;
    __bf16* vx     = (__bf16*)w; w += MT * 2;
    __bf16* ybf    = (__bf16*)w; w += MT * 2;
    __bf16* Wqx    = (__bf16*)w; w += WT * 2;
    __bf16* Wkx    = (__bf16*)w; w += WT * 2;
    __bf16* Wvx    = (__bf16*)w; w += WT * 2;
    __bf16* Wox    = (__bf16*)w; w += WT * 2;
    float*  pnum   = (float*)w;  w += NCHUNK * NCOL * 4;
    float*  pden   = (float*)w;  w += NCHUNK * NCOL * 4;
    float*  pbm    = (float*)w;  w += T_LEN * 4;
    float*  ml     = (float*)w;  w += DMODEL * 4;

    // 0) bf16 copies of GEMM operands
    f32_to_bf16<<<MT / 1024, 256, 0, stream>>>(query, qx);
    f32_to_bf16<<<MT / 1024, 256, 0, stream>>>(key_,  kx);
    f32_to_bf16<<<MT / 1024, 256, 0, stream>>>(value, vx);
    f32_to_bf16<<<WT / 1024, 256, 0, stream>>>(Wq, Wqx);
    f32_to_bf16<<<WT / 1024, 256, 0, stream>>>(Wk, Wkx);
    f32_to_bf16<<<WT / 1024, 256, 0, stream>>>(Wv, Wvx);
    f32_to_bf16<<<WT / 1024, 256, 0, stream>>>(Wo, Wox);

    // 1) pos_bias column max (ReLU folded)
    pb_colmax<<<T_LEN / 256, 256, 0, stream>>>(pb, pbm);

    // 2) q/k/v projections (WMMA + async LDS pipeline)
    dim3 gg(GN / BN, GM / BM);   // (8, 64)
    gemm_bf16_wmma<<<gg, 128, 0, stream>>>(qx, Wqx, bq, q);
    gemm_bf16_wmma<<<gg, 128, 0, stream>>>(kx, Wkx, bk, kbuf);
    gemm_bf16_wmma<<<gg, 128, 0, stream>>>(vx, Wvx, bv, vbuf);

    // 3) stabilization constant per d
    max_logit_kernel<<<DMODEL / 256, 256, 0, stream>>>(kbuf, pbm, ml);

    // 4) chunked scan for cs_num / cs_den
    dim3 sg(NCHUNK, NCOL / 256);  // (16, 4)
    scan_pass1<<<sg, 256, 0, stream>>>(kbuf, vbuf, ml, pnum, pden);
    scan_pass2<<<NCOL / 256, 256, 0, stream>>>(pnum, pden);
    scan_pass3<<<sg, 256, 0, stream>>>(kbuf, vbuf, ml, pnum, pden, cs_num, cs_den);

    // 5) AFT-local combine -> y (bf16)
    aft_combine<<<dim3(T_LEN, BATCH), DMODEL, 0, stream>>>(
        q, kbuf, vbuf, cs_num, cs_den, pb, ml, ybf);

    // 6) output projection (WMMA)
    gemm_bf16_wmma<<<gg, 128, 0, stream>>>(ybf, Wox, bo, out);
}
